// OneDFIDNet_45037027066530
// MI455X (gfx1250) — matrix operand order repaired and checked by reading, not compiled, not run on hardware
//
#include <hip/hip_runtime.h>
#include <hip/hip_bf16.h>

#define B_   4
#define T_   2048
#define CIN_ 5
#define F_   32
#define CF_  64
#define CH_  69
#define CR_  138
#define CRP_ 144
#define NP_  80
#define KREF_ (8*CRP_)   /* 1152 */
#define NL_  45
#define NREF_ 4
#define NSIG_ 276        /* B_*CH_ */
#define NSIGP_ 288

/* padded row counts (zero halos so fragment loads never need bounds checks) */
#define TH_  (T_ + 224)  /* h16: forward reach 7*maxdil = 224 */
#define TG_  (T_ + 8)    /* g16: forward reach 7 */
#define TP_  (T_ + 12)   /* post/refine: reach t-3 .. t+4 */
#define FP_  4           /* front pad rows for TP_ buffers */

typedef _Float16 h8_t  __attribute__((ext_vector_type(8)));
typedef _Float16 v16h  __attribute__((ext_vector_type(16)));
typedef float    v8f   __attribute__((ext_vector_type(8)));

__device__ __forceinline__ float act_tanh(float x) { return tanhf(x) * 0.98f + 0.02f * x; }
__device__ __forceinline__ float act_sig(float x)  { return 0.98f / (1.0f + expf(-x)) + 0.02f * x; }
__device__ __forceinline__ float act_relu(float x) { return x > 0.0f ? x : 0.02f * x; }

__device__ __forceinline__ v8f wmma16(v16h a, v16h b, v8f c) {
  return __builtin_amdgcn_wmma_f32_16x16x32_f16(false, a, false, b, (short)0, c, false, false);
}

// A fragment 16x32 f16: lane<16 holds K {0..15}, lane>=16 holds {16..31}
__device__ __forceinline__ v16h afragP(const _Float16* __restrict__ row, int c0, int hl) {
  union { v16h v; h8_t h[2]; } u;
  u.h[0] = *(const h8_t*)(row + c0 + hl * 16);
  u.h[1] = *(const h8_t*)(row + c0 + hl * 16 + 8);
  return u.v;
}

// B fragment 32x16 f16 from [N][K] storage: lane reads 16 contiguous K at wrow
__device__ __forceinline__ v16h bfrag(const _Float16* __restrict__ wrow) {
  union { v16h v; h8_t h[2]; } u;
  u.h[0] = *(const h8_t*)(wrow);
  u.h[1] = *(const h8_t*)(wrow + 8);
  return u.v;
}

// CDNA5 async global->LDS 16B copy (ASYNCcnt-tracked, no VGPR staging)
__device__ __forceinline__ void async_cp16(unsigned lds_off, const void* gaddr) {
  asm volatile("global_load_async_to_lds_b128 %0, %1, off"
               :: "v"(lds_off), "v"(gaddr) : "memory");
}
__device__ __forceinline__ void async_wait() {
  asm volatile("s_wait_asynccnt 0x0" ::: "memory");
}

// ---------------- weight prep ----------------
__global__ void k_prep_wave(const float* __restrict__ wy1, const float* __restrict__ wy2,
                            _Float16* __restrict__ wy1h, _Float16* __restrict__ wy2h) {
  int gid = blockIdx.x * blockDim.x + threadIdx.x;
  if (gid >= NL_ * 8 * CF_ * F_) return;
  int f = gid & 31, c = (gid >> 5) & 63, k = (gid >> 11) & 7, i = gid >> 14;
  size_t src = (((size_t)(i * 3 + 1) * 8 + k) * CF_ + c) * F_ + f;
  size_t dst = ((size_t)i * F_ + f) * 512 + k * 64 + c;
  wy1h[dst] = (_Float16)wy1[src];
  wy2h[dst] = (_Float16)wy2[src];
}

__global__ void k_prep_wz(const float* __restrict__ wz, _Float16* __restrict__ wzh) {
  int gid = blockIdx.x * blockDim.x + threadIdx.x;
  if (gid >= NL_ * 8 * F_ * CF_) return;
  int cp = gid & 63, f = (gid >> 6) & 31, k = (gid >> 11) & 7, i = gid >> 14;
  size_t src = (((size_t)(i * 3 + 1) * 8 + k) * F_ + f) * CF_ + cp;
  size_t dst = ((size_t)i * CF_ + cp) * 256 + k * 32 + f;
  wzh[dst] = (_Float16)wz[src];
}

__global__ void k_prep_post(const float* __restrict__ p0w, const float* __restrict__ p1w,
                            _Float16* __restrict__ postwh) {
  int gid = blockIdx.x * blockDim.x + threadIdx.x;
  if (gid >= 2 * 8 * CF_ * CF_) return;
  int co = gid & 63, ci = (gid >> 6) & 63, k = (gid >> 12) & 7, pp = gid >> 15;
  const float* src = pp ? p1w : p0w;
  size_t si = (((size_t)8 + k) * CF_ + ci) * CF_ + co;     // kh==1 slice
  size_t dst = ((size_t)pp * CF_ + co) * 512 + k * 64 + ci;
  postwh[dst] = (_Float16)src[si];
}

__global__ void k_prep_ref(const float* __restrict__ rtw, const float* __restrict__ rrw,
                           _Float16* __restrict__ refwh) {
  int gid = blockIdx.x * blockDim.x + threadIdx.x;
  if (gid >= NREF_ * 2 * NP_ * KREF_) return;
  int K = gid % KREF_;
  int rest = gid / KREF_;
  int n = rest % NP_;
  rest /= NP_;
  int conv = rest & 1;
  int i = rest >> 1;
  int c = K % CRP_, k = K / CRP_;
  float v = 0.0f;
  if (n < CH_ && c < CR_) {
    size_t si = (((size_t)i * 8 + k) * CR_ + c) * CH_ + n;
    v = conv ? rrw[si] : rtw[si];
  }
  refwh[gid] = (_Float16)v;
}

// DFT table: Dt[j][2n]=(-1)^n cos(pi*n*j/1024), Dt[j][2n+1]=(-1)^n sin(...)
__global__ void k_dft_table(_Float16* __restrict__ dft16) {
  int gid = blockIdx.x * blockDim.x + threadIdx.x;
  if (gid >= 1024 * 2048) return;
  int n = gid >> 11;
  int j = gid & 2047;
  long m = ((long)n * (long)j) & 2047;
  float th = (float)m * 0.00306796157577128245943836f; // pi/1024
  float sgn = (n & 1) ? -1.0f : 1.0f;
  float s, c;
  sincosf(th, &s, &c);
  _Float16* row = dft16 + (size_t)j * 2048;
  row[2 * n]     = (_Float16)(sgn * c);
  row[2 * n + 1] = (_Float16)(sgn * s);
}

// ---------------- init: h = dfh_tanh(x @ Wi + bi) ----------------
__global__ void k_init(const float* __restrict__ inp, const float* __restrict__ w,
                       const float* __restrict__ bias,
                       float* __restrict__ h32, _Float16* __restrict__ h16) {
  int gid = blockIdx.x * blockDim.x + threadIdx.x;
  if (gid >= B_ * T_) return;
  int b = gid / T_, t = gid % T_;
  float x[CIN_];
  for (int c = 0; c < CIN_; ++c) x[c] = inp[((size_t)b * CIN_ + c) * T_ + t];
  for (int f = 0; f < CF_; ++f) {
    float s = bias[f];
    for (int c = 0; c < CIN_; ++c) s += x[c] * w[c * CF_ + f];
    float v = act_tanh(s);
    h32[(size_t)gid * CF_ + f] = v;
    h16[((size_t)b * TH_ + t) * CF_ + f] = (_Float16)v;
  }
}

// ---------------- wave layer: gated conv (y1*y2 -> g) ----------------
__global__ __launch_bounds__(256) void k_wave_gate(
    const _Float16* __restrict__ h16,
    const _Float16* __restrict__ wy1h, const _Float16* __restrict__ wy2h,
    const float* __restrict__ by1, const float* __restrict__ by2,
    _Float16* __restrict__ g16, int layer, int dil) {
  __shared__ __align__(16) _Float16 sW[2 * F_ * 512];   // 64 KB: wy1|wy2 of this layer, [n][K]
  const _Float16* W1g = wy1h + (size_t)layer * F_ * 512;
  const _Float16* W2g = wy2h + (size_t)layer * F_ * 512;
  unsigned lds_base = (unsigned)(size_t)(void*)sW;
  for (int it = 0; it < 8; ++it) {
    int idx = threadIdx.x + it * 256;                   // 2048 chunks x 16B = 32KB
    async_cp16(lds_base + idx * 16, (const char*)W1g + (size_t)idx * 16);
    async_cp16(lds_base + 32768u + idx * 16, (const char*)W2g + (size_t)idx * 16);
  }
  async_wait();
  __syncthreads();

  int lane = threadIdx.x & 31;
  int hl = lane >> 4, lrow = lane & 15;
  int tile = blockIdx.x * 8 + (threadIdx.x >> 5);   // 512 tiles = B*T/16
  int b = tile >> 7;
  int t0 = (tile & 127) << 4;
  const _Float16* hb = h16 + (size_t)b * TH_ * CF_;
  int trow = t0 + lrow;

  v8f acc0 = {}, acc1 = {}, acc2 = {}, acc3 = {};
  for (int kk = 0; kk < 16; ++kk) {
    int k = kk >> 1, c0 = (kk & 1) * 32;
    int tr = trow + k * dil;                        // halo rows are zero, no check
    v16h a = afragP(hb + (size_t)tr * CF_, c0, hl);
    int k0 = kk * 32 + hl * 16;
    v16h m0 = bfrag(sW + (size_t)lrow * 512 + k0);
    v16h m1 = bfrag(sW + (size_t)(16 + lrow) * 512 + k0);
    v16h m2 = bfrag(sW + (size_t)(F_ + lrow) * 512 + k0);
    v16h m3 = bfrag(sW + (size_t)(F_ + 16 + lrow) * 512 + k0);
    acc0 = wmma16(a, m0, acc0);
    acc1 = wmma16(a, m1, acc1);
    acc2 = wmma16(a, m2, acc2);
    acc3 = wmma16(a, m3, acc3);
  }
  float bb0 = by1[layer * F_ + lrow],      bb1 = by1[layer * F_ + 16 + lrow];
  float bb2 = by2[layer * F_ + lrow],      bb3 = by2[layer * F_ + 16 + lrow];
  _Float16* gb = g16 + (size_t)b * TG_ * F_;
  for (int r = 0; r < 8; ++r) {
    int t = t0 + hl * 8 + r;
    float y1a = act_tanh(acc0[r] + bb0);
    float y1b = act_tanh(acc1[r] + bb1);
    float y2a = act_sig(acc2[r] + bb2);
    float y2b = act_sig(acc3[r] + bb3);
    _Float16* gp = gb + (size_t)t * F_;
    gp[lrow]      = (_Float16)(y1a * y2a);
    gp[16 + lrow] = (_Float16)(y1b * y2b);
  }
}

// ---------------- wave layer: z conv + residual (skip += z; h += z) ----------------
__global__ __launch_bounds__(256) void k_wave_z(
    const _Float16* __restrict__ g16, const _Float16* __restrict__ wzh,
    const float* __restrict__ bz,
    float* __restrict__ h32, _Float16* __restrict__ h16,
    float* __restrict__ skip32, int layer) {
  __shared__ __align__(16) _Float16 sW[CF_ * 256];   // 32 KB: wz[layer] in [n][K]
  const _Float16* Wg = wzh + (size_t)layer * CF_ * 256;
  unsigned lds_base = (unsigned)(size_t)(void*)sW;
  for (int it = 0; it < 8; ++it) {
    int idx = threadIdx.x + it * 256;
    async_cp16(lds_base + idx * 16, (const char*)Wg + (size_t)idx * 16);
  }
  async_wait();
  __syncthreads();

  int lane = threadIdx.x & 31;
  int hl = lane >> 4, lrow = lane & 15;
  int tile = blockIdx.x * 8 + (threadIdx.x >> 5);
  int b = tile >> 7;
  int t0 = (tile & 127) << 4;
  const _Float16* gb = g16 + (size_t)b * TG_ * F_;
  int trow = t0 + lrow;

  v8f acc[4] = {};
  for (int kk = 0; kk < 8; ++kk) {
    v16h a = afragP(gb + (size_t)(trow + kk) * F_, 0, hl);
    int k0 = kk * 32 + hl * 16;
#pragma unroll
    for (int ni = 0; ni < 4; ++ni) {
      v16h bm = bfrag(sW + (size_t)(ni * 16 + lrow) * 256 + k0);
      acc[ni] = wmma16(a, bm, acc[ni]);
    }
  }
  for (int ni = 0; ni < 4; ++ni) {
    int c = ni * 16 + lrow;
    float bzc = bz[layer * CF_ + c];
    for (int r = 0; r < 8; ++r) {
      int t = t0 + hl * 8 + r;
      size_t idx = ((size_t)b * T_ + t) * CF_ + c;
      float z = acc[ni][r] + bzc;
      skip32[idx] += z;
      float hn = h32[idx] + z;
      h32[idx] = hn;
      h16[((size_t)b * TH_ + t) * CF_ + c] = (_Float16)hn;
    }
  }
}

__global__ void k_tanhskip(const float* __restrict__ skip32, _Float16* __restrict__ a16) {
  int gid = blockIdx.x * blockDim.x + threadIdx.x;
  if (gid >= B_ * T_ * CF_) return;
  int c = gid % CF_;
  int bt = gid / CF_;
  int b = bt / T_, t = bt % T_;
  a16[((size_t)b * TP_ + FP_ + t) * CF_ + c] = (_Float16)act_tanh(skip32[gid]);
}

// ---------------- post conv (SAME, taps at t-3..t+4), mode 0=relu, 1=tanh ----------------
__global__ __launch_bounds__(256) void k_post(
    const _Float16* __restrict__ in16, const _Float16* __restrict__ pw,
    const float* __restrict__ pb, _Float16* __restrict__ out16, int mode) {
  __shared__ __align__(16) _Float16 sW[CF_ * 512];   // 64 KB
  unsigned lds_base = (unsigned)(size_t)(void*)sW;
  for (int it = 0; it < 16; ++it) {
    int idx = threadIdx.x + it * 256;                // 4096 chunks x 16B
    async_cp16(lds_base + idx * 16, (const char*)pw + (size_t)idx * 16);
  }
  async_wait();
  __syncthreads();

  int lane = threadIdx.x & 31;
  int hl = lane >> 4, lrow = lane & 15;
  int tile = blockIdx.x * 8 + (threadIdx.x >> 5);
  int b = tile >> 7;
  int t0 = (tile & 127) << 4;
  const _Float16* ib = in16 + ((size_t)b * TP_ + FP_) * CF_;
  int trow = t0 + lrow;

  v8f acc[4] = {};
  for (int kk = 0; kk < 16; ++kk) {
    int k = kk >> 1, c0 = (kk & 1) * 32;
    v16h a = afragP(ib + (size_t)(trow + k - 3) * CF_, c0, hl);
    int k0 = kk * 32 + hl * 16;
#pragma unroll
    for (int ni = 0; ni < 4; ++ni) {
      v16h bm = bfrag(sW + (size_t)(ni * 16 + lrow) * 512 + k0);
      acc[ni] = wmma16(a, bm, acc[ni]);
    }
  }
  for (int ni = 0; ni < 4; ++ni) {
    int c = ni * 16 + lrow;
    float bc = pb[c];
    for (int r = 0; r < 8; ++r) {
      int t = t0 + hl * 8 + r;
      float v = acc[ni][r] + bc;
      v = mode ? act_tanh(v) : act_relu(v);
      out16[((size_t)b * TP_ + FP_ + t) * CF_ + c] = (_Float16)v;
    }
  }
}

// ---------------- DFT input matrix A[s][t] ----------------
__global__ void k_dftA(const _Float16* __restrict__ q16, const float* __restrict__ inp,
                       _Float16* __restrict__ dftA) {
  int gid = blockIdx.x * blockDim.x + threadIdx.x;
  if (gid >= NSIG_ * T_) return;
  int s = gid / T_, t = gid % T_;
  int b = s / CH_, ch = s % CH_;
  float v;
  if (ch < CF_) v = (float)q16[((size_t)b * TP_ + FP_ + t) * CF_ + ch];
  else          v = inp[((size_t)b * CIN_ + (ch - CF_)) * T_ + t];
  if (t < 2) v *= 0.5f;
  dftA[(size_t)s * T_ + t] = (_Float16)v;
}

// ---------------- DFT GEMM: hout[s][j] = sum_K A[s][K]*Dt[j][K] ----------------
__global__ __launch_bounds__(256) void k_dft_gemm(
    const _Float16* __restrict__ A, const _Float16* __restrict__ Dt,
    const float* __restrict__ scal, float* __restrict__ hout) {
  int lane = threadIdx.x & 31;
  int hl = lane >> 4, lrow = lane & 15;
  int tile = blockIdx.x * 8 + (threadIdx.x >> 5);   // 18*128 = 2304 tiles
  int mi = tile >> 7;
  int ni = tile & 127;
  int s0 = mi * 16, j0 = ni * 16;
  const _Float16* arow = A + (size_t)(s0 + lrow) * T_;
  const _Float16* brow = Dt + (size_t)(j0 + lrow) * T_;
  v8f acc = {};
  for (int kk = 0; kk < 64; ++kk) {
    int c0 = kk * 32;
    v16h a = afragP(arow, c0, hl);
    v16h bm = bfrag(brow + c0 + hl * 16);
    acc = wmma16(a, bm, acc);
  }
  int j = j0 + lrow;
  for (int r = 0; r < 8; ++r) {
    int s = s0 + hl * 8 + r;
    if (s < NSIG_) {
      int b = s / CH_, ch = s % CH_;
      float v = acc[r];
      if (ch >= CF_) v *= scal[b];
      hout[((size_t)b * T_ + j) * CH_ + ch] = v;
    }
  }
}

__global__ void k_fnorm(const float* __restrict__ hout, unsigned* __restrict__ fn) {
  int gid = blockIdx.x * blockDim.x + threadIdx.x;
  if (gid >= B_ * T_ * CH_) return;
  int b = gid / (T_ * CH_);
  atomicMax(&fn[b], __float_as_uint(fabsf(hout[gid])));
}

__global__ void k_refinit(const float* __restrict__ hout, const unsigned* __restrict__ fn,
                          float* __restrict__ r32, _Float16* __restrict__ r16) {
  int gid = blockIdx.x * blockDim.x + threadIdx.x;
  if (gid >= B_ * T_ * CRP_) return;
  int c = gid % CRP_;
  int bt = gid / CRP_;
  int b = bt / T_, t = bt % T_;
  float v = 0.0f;
  if (c < CR_) {
    float fnv = __uint_as_float(fn[b]);
    v = hout[(size_t)bt * CH_ + (c % CH_)] / fnv;
    r32[(size_t)bt * CR_ + c] = v;
  }
  r16[((size_t)b * TP_ + FP_ + t) * CRP_ + c] = (_Float16)v;
}

// ---------------- refine step: tt/rr convs + residual concat ----------------
__global__ __launch_bounds__(256) void k_refine(
    const _Float16* __restrict__ rin16, const float* __restrict__ rin32,
    float* __restrict__ rout32, _Float16* __restrict__ rout16,
    const _Float16* __restrict__ W,      // [2][NP][KREF] (t-conv then r-conv)
    const float* __restrict__ btb, const float* __restrict__ brb) {
  int lane = threadIdx.x & 31;
  int hl = lane >> 4, lrow = lane & 15;
  int tile = blockIdx.x * 8 + (threadIdx.x >> 5);
  int b = tile >> 7;
  int t0 = (tile & 127) << 4;
  const _Float16* rb = rin16 + ((size_t)b * TP_ + FP_) * CRP_;
  int trow = t0 + lrow;

  v8f acc[10] = {};
  for (int kk = 0; kk < 36; ++kk) {
    int kb = kk * 32;
    union { v16h v; h8_t h[2]; } ua;
    {
      int K0 = kb + hl * 16;
      int k = K0 / CRP_, c = K0 % CRP_;       // 8-chunks never straddle a tap (144 % 8 == 0)
      ua.h[0] = *(const h8_t*)(rb + (size_t)(trow + k - 3) * CRP_ + c);
    }
    {
      int K0 = kb + hl * 16 + 8;
      int k = K0 / CRP_, c = K0 % CRP_;
      ua.h[1] = *(const h8_t*)(rb + (size_t)(trow + k - 3) * CRP_ + c);
    }
    int k0 = kb + hl * 16;
    __builtin_prefetch(W + (size_t)lrow * KREF_ + k0 + 512, 0, 1);
#pragma unroll
    for (int ni = 0; ni < 5; ++ni) {
      v16h wt = bfrag(W + (size_t)(ni * 16 + lrow) * KREF_ + k0);
      acc[ni] = wmma16(ua.v, wt, acc[ni]);
      v16h wr = bfrag(W + (size_t)(NP_ + ni * 16 + lrow) * KREF_ + k0);
      acc[5 + ni] = wmma16(ua.v, wr, acc[5 + ni]);
    }
  }
  for (int ni = 0; ni < 5; ++ni) {
    int n = ni * 16 + lrow;
    if (n < CH_) {
      float bt_v = btb[n], br_v = brb[n];
      for (int r = 0; r < 8; ++r) {
        int t = t0 + hl * 8 + r;
        size_t b32 = ((size_t)b * T_ + t) * CR_;
        size_t b16 = ((size_t)b * TP_ + FP_ + t) * CRP_;
        float tv = act_tanh(acc[ni][r] + bt_v) + rin32[b32 + n];
        rout32[b32 + n] = tv;
        rout16[b16 + n] = (_Float16)tv;
        float rv = act_relu(acc[5 + ni][r] + br_v) + rin32[b32 + CH_ + n];
        rout32[b32 + CH_ + n] = rv;
        rout16[b16 + CH_ + n] = (_Float16)rv;
      }
    }
  }
}

// ---------------- final dense + output scaling ----------------
__global__ void k_final(const float* __restrict__ ref32, const float* __restrict__ hout,
                        const unsigned* __restrict__ fn, const float* __restrict__ scal,
                        const float* __restrict__ dw, const float* __restrict__ db,
                        float* __restrict__ out) {
  int gid = blockIdx.x * blockDim.x + threadIdx.x;
  if (gid >= B_ * T_) return;
  int b = gid / T_;
  float ms = scal[b];
  float fnv = __uint_as_float(fn[b]);
  float s0 = db[0], s1 = db[1];
  const float* rp = ref32 + (size_t)gid * CR_;
  for (int c = 0; c < CR_; ++c) {
    float v = rp[c] * fnv * ms;
    s0 += v * dw[2 * c];
    s1 += v * dw[2 * c + 1];
  }
  const float* hp = hout + (size_t)gid * CH_;
  float m2 = 2.0f - ms;
  for (int c = 0; c < CH_; ++c) {
    float v = m2 * hp[c];
    s0 += v * dw[2 * (CR_ + c)];
    s1 += v * dw[2 * (CR_ + c) + 1];
  }
  out[2 * (size_t)gid]     = 5.0f * s0;
  out[2 * (size_t)gid + 1] = s1 / fnv;
}

extern "C" void kernel_launch(void* const* d_in, const int* in_sizes, int n_in,
                              void* d_out, int out_size, void* d_ws, size_t ws_size,
                              hipStream_t stream) {
  (void)in_sizes; (void)n_in; (void)out_size; (void)ws_size;
  const float* inp  = (const float*)d_in[1];
  const float* scal = (const float*)d_in[2];
  const float* diw  = (const float*)d_in[3];
  const float* dib  = (const float*)d_in[4];
  const float* wy1  = (const float*)d_in[5];
  const float* by1  = (const float*)d_in[6];
  const float* wy2  = (const float*)d_in[7];
  const float* by2  = (const float*)d_in[8];
  const float* wz   = (const float*)d_in[9];
  const float* bz   = (const float*)d_in[10];
  const float* p0w  = (const float*)d_in[11];
  const float* p0b  = (const float*)d_in[12];
  const float* p1w  = (const float*)d_in[13];
  const float* p1b  = (const float*)d_in[14];
  const float* rrw  = (const float*)d_in[15];
  const float* rrb  = (const float*)d_in[16];
  const float* rtw  = (const float*)d_in[17];
  const float* rtb  = (const float*)d_in[18];
  const float* dw   = (const float*)d_in[19];
  const float* db   = (const float*)d_in[20];
  float* out = (float*)d_out;

  char* p = (char*)d_ws;
  auto alloc = [&](size_t bytes) -> char* {
    char* r = p; p += (bytes + 255) & ~(size_t)255; return r;
  };
  float*    h32    = (float*)alloc((size_t)B_ * T_ * CF_ * 4);
  _Float16* h16    = (_Float16*)alloc((size_t)B_ * TH_ * CF_ * 2);
  float*    skip32 = (float*)alloc((size_t)B_ * T_ * CF_ * 4);
  _Float16* g16    = (_Float16*)alloc((size_t)B_ * TG_ * F_ * 2);
  _Float16* a16    = (_Float16*)alloc((size_t)B_ * TP_ * CF_ * 2);
  _Float16* p16    = (_Float16*)alloc((size_t)B_ * TP_ * CF_ * 2);
  _Float16* q16    = (_Float16*)alloc((size_t)B_ * TP_ * CF_ * 2);
  _Float16* wy1h   = (_Float16*)alloc((size_t)NL_ * F_ * 512 * 2);
  _Float16* wy2h   = (_Float16*)alloc((size_t)NL_ * F_ * 512 * 2);
  _Float16* wzh    = (_Float16*)alloc((size_t)NL_ * CF_ * 256 * 2);
  _Float16* postwh = (_Float16*)alloc((size_t)2 * CF_ * 512 * 2);
  _Float16* refwh  = (_Float16*)alloc((size_t)NREF_ * 2 * NP_ * KREF_ * 2);
  _Float16* dft16  = (_Float16*)alloc((size_t)T_ * T_ * 2);
  _Float16* dftA   = (_Float16*)alloc((size_t)NSIGP_ * T_ * 2);
  float*    hout   = (float*)alloc((size_t)B_ * T_ * CH_ * 4);
  unsigned* fn     = (unsigned*)alloc(256);
  float*    r32a   = (float*)alloc((size_t)B_ * T_ * CR_ * 4);
  float*    r32b   = (float*)alloc((size_t)B_ * T_ * CR_ * 4);
  _Float16* r16a   = (_Float16*)alloc((size_t)B_ * TP_ * CRP_ * 2);
  _Float16* r16b   = (_Float16*)alloc((size_t)B_ * TP_ * CRP_ * 2);

  hipMemsetAsync(skip32, 0, (size_t)B_ * T_ * CF_ * 4, stream);
  hipMemsetAsync(fn, 0, 256, stream);
  hipMemsetAsync(h16, 0, (size_t)B_ * TH_ * CF_ * 2, stream);
  hipMemsetAsync(g16, 0, (size_t)B_ * TG_ * F_ * 2, stream);
  hipMemsetAsync(a16, 0, (size_t)B_ * TP_ * CF_ * 2, stream);
  hipMemsetAsync(p16, 0, (size_t)B_ * TP_ * CF_ * 2, stream);
  hipMemsetAsync(dftA, 0, (size_t)NSIGP_ * T_ * 2, stream);
  hipMemsetAsync(r16a, 0, (size_t)B_ * TP_ * CRP_ * 2, stream);
  hipMemsetAsync(r16b, 0, (size_t)B_ * TP_ * CRP_ * 2, stream);

  k_prep_wave<<<(NL_ * 8 * CF_ * F_ + 255) / 256, 256, 0, stream>>>(wy1, wy2, wy1h, wy2h);
  k_prep_wz<<<(NL_ * 8 * F_ * CF_ + 255) / 256, 256, 0, stream>>>(wz, wzh);
  k_prep_post<<<(2 * 8 * CF_ * CF_ + 255) / 256, 256, 0, stream>>>(p0w, p1w, postwh);
  k_prep_ref<<<(NREF_ * 2 * NP_ * KREF_ + 255) / 256, 256, 0, stream>>>(rtw, rrw, refwh);
  k_dft_table<<<(1024 * 2048 + 255) / 256, 256, 0, stream>>>(dft16);

  k_init<<<(B_ * T_ + 255) / 256, 256, 0, stream>>>(inp, diw, dib, h32, h16);

  static const int DILS[15] = {1, 2, 3, 4, 6, 8, 10, 12, 14, 16, 18, 20, 24, 28, 32};
  for (int i = 0; i < NL_; ++i) {
    int dil = DILS[i % 15];
    k_wave_gate<<<64, 256, 0, stream>>>(h16, wy1h, wy2h, by1, by2, g16, i, dil);
    k_wave_z<<<64, 256, 0, stream>>>(g16, wzh, bz, h32, h16, skip32, i);
  }
  k_tanhskip<<<(B_ * T_ * CF_ + 255) / 256, 256, 0, stream>>>(skip32, a16);
  k_post<<<64, 256, 0, stream>>>(a16, postwh, p0b, p16, 0);
  k_post<<<64, 256, 0, stream>>>(p16, postwh + (size_t)CF_ * 512, p1b, q16, 1);
  k_dftA<<<(NSIG_ * T_ + 255) / 256, 256, 0, stream>>>(q16, inp, dftA);
  k_dft_gemm<<<288, 256, 0, stream>>>(dftA, dft16, scal, hout);
  k_fnorm<<<(B_ * T_ * CH_ + 255) / 256, 256, 0, stream>>>(hout, fn);
  k_refinit<<<(B_ * T_ * CRP_ + 255) / 256, 256, 0, stream>>>(hout, fn, r32a, r16a);
  for (int i = 0; i < NREF_; ++i) {
    const float* btb = rtb + (size_t)i * CH_;
    const float* brb = rrb + (size_t)i * CH_;
    const _Float16* W = refwh + (size_t)i * 2 * NP_ * KREF_;
    if ((i & 1) == 0)
      k_refine<<<64, 256, 0, stream>>>(r16a, r32a, r32b, r16b, W, btb, brb);
    else
      k_refine<<<64, 256, 0, stream>>>(r16b, r32b, r32a, r16a, W, btb, brb);
  }
  k_final<<<(B_ * T_ + 255) / 256, 256, 0, stream>>>(r32a, hout, fn, scal, dw, db, out);
}
